// AGCRNCell_14096082665701
// MI455X (gfx1250) — compile-verified, matrix operand order checked by reading, not run on hardware
//
#include <hip/hip_runtime.h>
#include <hip/hip_bf16.h>

typedef __attribute__((ext_vector_type(16))) __bf16 v16bf;
typedef __attribute__((ext_vector_type(8)))  __bf16 v8bf;
typedef __attribute__((ext_vector_type(8)))  float  v8f;
typedef unsigned int v4u __attribute__((ext_vector_type(4)));
typedef int          v4i __attribute__((ext_vector_type(4)));
typedef int          v8i __attribute__((ext_vector_type(8)));

#define NODES 2048
#define BATCH 32
#define CIN   66          // dim_in(2) + hidden(64)
#define NJ    (BATCH*CIN) // 2112 packed batch*channel columns
#define HID   64

// Tensor Data Mover availability (device pass only; host pass -> fallback)
#if defined(__AMDGCN__) && __has_builtin(__builtin_amdgcn_tensor_load_to_lds)
#define USE_TDM 1
#else
#define USE_TDM 0
#endif

#if USE_TDM
#if __clang_major__ >= 23
#define TDM_LOAD(g0, g1) __builtin_amdgcn_tensor_load_to_lds((g0), (g1), z4, z4, z8, 0)
#else
#define TDM_LOAD(g0, g1) __builtin_amdgcn_tensor_load_to_lds((g0), (g1), z4, z4, 0)
#endif
#if __has_builtin(__builtin_amdgcn_s_wait_tensorcnt)
#define TDM_WAIT0() __builtin_amdgcn_s_wait_tensorcnt((short)0)
#else
#define TDM_WAIT0() asm volatile("s_wait_tensorcnt 0x0" ::: "memory")
#endif

// D# group 0: count=1 | lds byte addr | 57-bit global byte addr | type=2
__device__ inline v4u tdm_g0(unsigned int lds_off, const void* gaddr) {
  const unsigned long long ga = (unsigned long long)(uintptr_t)gaddr;
  v4u g;
  g[0] = 1u;                                   // count=1, user mode
  g[1] = lds_off;                              // lds_addr (bytes)
  g[2] = (unsigned int)ga;                     // global_addr[31:0]
  g[3] = (unsigned int)((ga >> 32) & 0x01FFFFFFu) | (2u << 30);  // addr[56:32]|type=2
  return g;
}
// D# group 1 for a 2D bf16 tile with LDS padding 128B+16B (stride-72 elems):
//   data_size=2B, pad_enable, pad_interval=32 DWORDs (code 4),
//   pad_amount=4 DWORDs (code 3)
__device__ inline v8i tdm_g1_2d(unsigned int td0, unsigned int td1,
                                unsigned int tl0, unsigned int tl1,
                                unsigned int stride0) {
  v8i g;
  g[0] = (int)((1u << 16) | (1u << 20) | (4u << 22) | (3u << 25));
  g[1] = (int)(td0 << 16);                     // tensor_dim0[15:0] @ bits63:48
  g[2] = (int)((td0 >> 16) | (td1 << 16));     // dim0[31:16] | dim1[15:0]
  g[3] = (int)((td1 >> 16) | (tl0 << 16));     // dim1[31:16] | tile_dim0
  g[4] = (int)tl1;                             // tile_dim1 (tile_dim2=0)
  g[5] = (int)stride0;                         // tensor_dim0_stride[31:0]
  g[6] = 0;                                    // stride0[47:32], stride1 lo
  g[7] = 0;
  return g;
}
#endif  // USE_TDM

// ---------------------------------------------------------------------------
// Kernel 1: supports = softmax(relu(E E^T), axis=1) -> bf16 [N][N]
// ---------------------------------------------------------------------------
__global__ __launch_bounds__(256) void supports_kernel(
    const float* __restrict__ E, __bf16* __restrict__ S) {
  const int n = blockIdx.x;
  const int t = threadIdx.x;
  __shared__ float se[10];
  __shared__ float red[256];
  if (t < 10) se[t] = E[n * 10 + t];
  __syncthreads();

  float v[8];
  float mx = 0.0f;  // relu output >= 0
#pragma unroll
  for (int i = 0; i < 8; ++i) {
    const int m = t + 256 * i;
    float d = 0.0f;
#pragma unroll
    for (int c = 0; c < 10; ++c) d += se[c] * E[m * 10 + c];
    v[i] = d > 0.0f ? d : 0.0f;
    mx = fmaxf(mx, v[i]);
  }
  red[t] = mx;
  __syncthreads();
  for (int s = 128; s > 0; s >>= 1) {
    if (t < s) red[t] = fmaxf(red[t], red[t + s]);
    __syncthreads();
  }
  mx = red[0];
  __syncthreads();

  float sum = 0.0f;
#pragma unroll
  for (int i = 0; i < 8; ++i) {
    v[i] = __expf(v[i] - mx);
    sum += v[i];
  }
  red[t] = sum;
  __syncthreads();
  for (int s = 128; s > 0; s >>= 1) {
    if (t < s) red[t] += red[t + s];
    __syncthreads();
  }
  const float inv = 1.0f / red[0];
#pragma unroll
  for (int i = 0; i < 8; ++i)
    S[(size_t)n * NODES + t + 256 * i] = (__bf16)(v[i] * inv);
}

// ---------------------------------------------------------------------------
// Kernel 2: pack X0^T[j = b*66+c][m] = concat(x, state)  (bf16, m contiguous)
// ---------------------------------------------------------------------------
__global__ __launch_bounds__(256) void pack_kernel(
    const float* __restrict__ x, const float* __restrict__ state,
    __bf16* __restrict__ X0T) {
  const int mbase = blockIdx.x * 32;
  const int b = blockIdx.y;
  const int t = threadIdx.x;
  __shared__ __bf16 tile[CIN * 33];
  for (int off = t; off < 32 * CIN; off += 256) {
    const int c = off % CIN, mi = off / CIN;
    const size_t bm = (size_t)b * NODES + mbase + mi;
    const float val = (c < 2) ? x[bm * 2 + c] : state[bm * HID + (c - 2)];
    tile[c * 33 + mi] = (__bf16)val;
  }
  __syncthreads();
  for (int off = t; off < CIN * 32; off += 256) {
    const int c = off / 32, mi = off % 32;
    X0T[(size_t)(b * CIN + c) * NODES + mbase + mi] = tile[c * 33 + mi];
  }
}

// ---------------------------------------------------------------------------
// WMMA bf16 NT-GEMM:  Out^T[j][n] = sum_m A[n][m] * B^T[j][m]
//   EPI==0: Out = A*B        EPI==1: Out = 2*A*B - Xp   (Xp same layout)
// Block 128(M)x64(J), 8 waves (4x2), wave tile 32x32 = 2x2 WMMA, K-step 64.
// TDM path: double-buffered LDS tiles fetched by tensor_load_to_lds (wave 0),
// padded to stride-72 by the D# pad fields; 1 barrier / iteration.
// ---------------------------------------------------------------------------
#define BK 64
#define LDA 72

union Frag { v16bf v; v8bf h[2]; };

template <int EPI>
__device__ inline void epi_store(__bf16* __restrict__ Out,
                                 const __bf16* __restrict__ Xp, v8f acc,
                                 int j, int nb) {
  if (EPI) {
    v8bf xp = *(const v8bf*)&Xp[(size_t)j * NODES + nb];
#pragma unroll
    for (int e = 0; e < 8; ++e) acc[e] = 2.0f * acc[e] - (float)xp[e];
  }
  v8bf o;
#pragma unroll
  for (int e = 0; e < 8; ++e) o[e] = (__bf16)acc[e];
  *(v8bf*)&Out[(size_t)j * NODES + nb] = o;
}

template <int EPI>
__global__ __launch_bounds__(256) void gemm_nt(
    const __bf16* __restrict__ A, const __bf16* __restrict__ Bm,
    const __bf16* __restrict__ Xp, __bf16* __restrict__ Out) {
#if USE_TDM
  __shared__ __bf16 As[2 * 128 * LDA];
  __shared__ __bf16 Bs[2 * 64 * LDA];
#else
  __shared__ __bf16 As[128 * LDA];
  __shared__ __bf16 Bs[64 * LDA];
#endif
  const int t = threadIdx.x, lane = t & 31, wave = t >> 5;
  const int waveM = wave & 3, waveN = wave >> 2;
  const int rowBase = blockIdx.x * 128, colBase = blockIdx.y * 64;

  v8f acc00 = {}, acc01 = {}, acc10 = {}, acc11 = {};

#if USE_TDM
  const unsigned int asOff = (unsigned int)(uintptr_t)(void*)&As[0];
  const unsigned int bsOff = (unsigned int)(uintptr_t)(void*)&Bs[0];
  v4i z4 = {};
  v8i z8 = {};
  (void)z8;
  const v8i g1A = tdm_g1_2d(NODES, NODES, BK, 128, NODES);
  const v8i g1B = tdm_g1_2d(NODES, NJ, BK, 64, NODES);
  if (wave == 0) {  // prefetch tile 0 into buffer 0
    TDM_LOAD(tdm_g0(asOff, &A[(size_t)rowBase * NODES]), g1A);
    TDM_LOAD(tdm_g0(bsOff, &Bm[(size_t)colBase * NODES]), g1B);
  }
  for (int it = 0; it < NODES / BK; ++it) {
    if (wave == 0) TDM_WAIT0();   // current tile landed in LDS
    __syncthreads();              // publish; all readers done with other buf
    if (wave == 0 && it + 1 < NODES / BK) {
      const int kb = (it + 1) * BK;
      const unsigned int ab = asOff + (unsigned int)(((it + 1) & 1) * (128 * LDA * 2));
      const unsigned int bb = bsOff + (unsigned int)(((it + 1) & 1) * (64 * LDA * 2));
      TDM_LOAD(tdm_g0(ab, &A[(size_t)rowBase * NODES + kb]), g1A);
      TDM_LOAD(tdm_g0(bb, &Bm[(size_t)colBase * NODES + kb]), g1B);
    }
    const int aB = (it & 1) * (128 * LDA);
    const int bB = (it & 1) * (64 * LDA);
#else
  for (int kb = 0; kb < NODES; kb += BK) {
#pragma unroll
    for (int r = 0; r < 4; ++r) {  // A tile: 128 rows x 64 k
      const int ch = t + 256 * r;
      const int row = ch >> 3, ko = (ch & 7) << 3;
      *(v8bf*)&As[row * LDA + ko] =
          *(const v8bf*)&A[(size_t)(rowBase + row) * NODES + kb + ko];
    }
#pragma unroll
    for (int r = 0; r < 2; ++r) {  // B tile: 64 cols x 64 k
      const int ch = t + 256 * r;
      const int col = ch >> 3, ko = (ch & 7) << 3;
      *(v8bf*)&Bs[col * LDA + ko] =
          *(const v8bf*)&Bm[(size_t)(colBase + col) * NODES + kb + ko];
    }
    __syncthreads();
    const int aB = 0;
    const int bB = 0;
#endif
#pragma unroll
    for (int ks = 0; ks < 2; ++ks) {
      const int kr = ks * 32 + ((lane >> 4) << 3);
      Frag a0, a1, b0, b1;
      const int row0 = aB + (waveM * 32 + (lane & 15)) * LDA;
      a0.h[0] = *(const v8bf*)&As[row0 + kr];
      a0.h[1] = *(const v8bf*)&As[row0 + kr + 16];
      a1.h[0] = *(const v8bf*)&As[row0 + 16 * LDA + kr];
      a1.h[1] = *(const v8bf*)&As[row0 + 16 * LDA + kr + 16];
      const int col0 = bB + (waveN * 32 + (lane & 15)) * LDA;
      b0.h[0] = *(const v8bf*)&Bs[col0 + kr];
      b0.h[1] = *(const v8bf*)&Bs[col0 + kr + 16];
      b1.h[0] = *(const v8bf*)&Bs[col0 + 16 * LDA + kr];
      b1.h[1] = *(const v8bf*)&Bs[col0 + 16 * LDA + kr + 16];
      acc00 = __builtin_amdgcn_wmma_f32_16x16x32_bf16(false, a0.v, false, b0.v,
                                                      (short)0, acc00, false, false);
      acc01 = __builtin_amdgcn_wmma_f32_16x16x32_bf16(false, a0.v, false, b1.v,
                                                      (short)0, acc01, false, false);
      acc10 = __builtin_amdgcn_wmma_f32_16x16x32_bf16(false, a1.v, false, b0.v,
                                                      (short)0, acc10, false, false);
      acc11 = __builtin_amdgcn_wmma_f32_16x16x32_bf16(false, a1.v, false, b1.v,
                                                      (short)0, acc11, false, false);
    }
#if !USE_TDM
    __syncthreads();
#endif
  }

  const int nb0 = rowBase + waveM * 32 + ((lane >> 4) << 3);
  const int j0 = colBase + waveN * 32 + (lane & 15);
  epi_store<EPI>(Out, Xp, acc00, j0, nb0);
  epi_store<EPI>(Out, Xp, acc01, j0 + 16, nb0);
  epi_store<EPI>(Out, Xp, acc10, j0, nb0 + 16);
  epi_store<EPI>(Out, Xp, acc11, j0 + 16, nb0 + 16);
}

// ---------------------------------------------------------------------------
// Kernel: gate  zr = sigmoid(x_g W_gate + b); save r; pack candidate into C0T
// (C0T aliases X0T; each block only writes its own column n -> race-free)
// ---------------------------------------------------------------------------
__global__ __launch_bounds__(128) void gate_kernel(
    const __bf16* __restrict__ X0T, const __bf16* __restrict__ X1T,
    const __bf16* __restrict__ X2T, const float* __restrict__ Wg,
    const float* __restrict__ bg, const float* __restrict__ x,
    const float* __restrict__ state, __bf16* __restrict__ C0T,
    float* __restrict__ r_ws) {
  const int n = blockIdx.x, b = blockIdx.y, t = threadIdx.x;
  __shared__ float xg[3 * CIN];
  for (int i = t; i < 3 * CIN; i += 128) {
    const int k = i / CIN, c = i % CIN;
    const __bf16* Xk = (k == 0) ? X0T : (k == 1) ? X1T : X2T;
    xg[i] = (float)Xk[(size_t)(b * CIN + c) * NODES + n];
  }
  __syncthreads();
  float acc = bg[t];
  for (int i = 0; i < 3 * CIN; ++i) acc += xg[i] * Wg[i * 128 + t];
  const float zr = 1.0f / (1.0f + __expf(-acc));
  const size_t bn = (size_t)b * NODES + n;
  if (t < HID) {  // z half -> candidate channels 2..65
    C0T[(size_t)(b * CIN + t + 2) * NODES + n] = (__bf16)(zr * state[bn * HID + t]);
  } else {        // r half
    r_ws[bn * HID + (t - HID)] = zr;
    if (t < HID + 2)  // candidate channels 0..1 = x
      C0T[(size_t)(b * CIN + (t - HID)) * NODES + n] = (__bf16)x[bn * 2 + (t - HID)];
  }
}

// ---------------------------------------------------------------------------
// Kernel: final  hc = tanh(c_g W_update + b);  h = r*state + (1-r)*hc
// ---------------------------------------------------------------------------
__global__ __launch_bounds__(64) void final_kernel(
    const __bf16* __restrict__ C0T, const __bf16* __restrict__ C1T,
    const __bf16* __restrict__ C2T, const float* __restrict__ Wu,
    const float* __restrict__ bu, const float* __restrict__ r_ws,
    const float* __restrict__ state, float* __restrict__ out) {
  const int n = blockIdx.x, b = blockIdx.y, t = threadIdx.x;
  __shared__ float cg[3 * CIN];
  for (int i = t; i < 3 * CIN; i += 64) {
    const int k = i / CIN, c = i % CIN;
    const __bf16* Ck = (k == 0) ? C0T : (k == 1) ? C1T : C2T;
    cg[i] = (float)Ck[(size_t)(b * CIN + c) * NODES + n];
  }
  __syncthreads();
  float acc = bu[t];
  for (int i = 0; i < 3 * CIN; ++i) acc += cg[i] * Wu[i * HID + t];
  const float hc = tanhf(acc);
  const size_t bn = (size_t)b * NODES + n;
  const float r = r_ws[bn * HID + t];
  out[bn * HID + t] = r * state[bn * HID + t] + (1.0f - r) * hc;
}

// ---------------------------------------------------------------------------
extern "C" void kernel_launch(void* const* d_in, const int* in_sizes, int n_in,
                              void* d_out, int out_size, void* d_ws,
                              size_t ws_size, hipStream_t stream) {
  (void)in_sizes; (void)n_in; (void)out_size; (void)ws_size;
  const float* x     = (const float*)d_in[0];
  const float* state = (const float*)d_in[1];
  const float* E     = (const float*)d_in[2];
  const float* Wg    = (const float*)d_in[3];
  const float* bg    = (const float*)d_in[4];
  const float* Wu    = (const float*)d_in[5];
  const float* bu    = (const float*)d_in[6];
  float* out = (float*)d_out;

  char* ws = (char*)d_ws;
  const size_t SZ_S = (size_t)NODES * NODES * 2;  // 8 MB bf16
  const size_t SZ_B = (size_t)NJ * NODES * 2;     // 8.65 MB bf16
  __bf16* S  = (__bf16*)(ws);
  __bf16* B0 = (__bf16*)(ws + SZ_S);
  __bf16* B1 = (__bf16*)(ws + SZ_S + SZ_B);
  __bf16* B2 = (__bf16*)(ws + SZ_S + 2 * SZ_B);
  float*  rw = (float*)(ws + SZ_S + 3 * SZ_B);

  const dim3 gGemm(NODES / 128, NJ / 64);  // 16 x 33

  supports_kernel<<<NODES, 256, 0, stream>>>(E, S);
  pack_kernel<<<dim3(NODES / 32, BATCH), 256, 0, stream>>>(x, state, B0);
  gemm_nt<0><<<gGemm, 256, 0, stream>>>(S, B0, nullptr, B1);  // X1 = S X0
  gemm_nt<1><<<gGemm, 256, 0, stream>>>(S, B1, B0, B2);       // X2 = 2 S X1 - X0
  gate_kernel<<<dim3(NODES, BATCH), 128, 0, stream>>>(B0, B1, B2, Wg, bg, x,
                                                      state, B0, rw);
  gemm_nt<0><<<gGemm, 256, 0, stream>>>(S, B0, nullptr, B1);  // C1 = S C0
  gemm_nt<1><<<gGemm, 256, 0, stream>>>(S, B1, B0, B2);       // C2 = 2 S C1 - C0
  final_kernel<<<dim3(NODES, BATCH), 64, 0, stream>>>(B0, B1, B2, Wu, bu, rw,
                                                      state, out);
}
